// LowRankAttention_80427557585639
// MI455X (gfx1250) — compile-verified, matrix operand order checked by reading, not run on hardware
//
#include <hip/hip_runtime.h>

typedef __bf16 bf16_t;
typedef __attribute__((ext_vector_type(16))) __bf16 v16bf;
typedef __attribute__((ext_vector_type(8)))  __bf16 v8bf;
typedef __attribute__((ext_vector_type(8)))  float  v8f;

#define B_    4
#define N_    196
#define C_    13
#define D_    768
#define H_    12
#define HD_   64
#define BH_   (B_*H_)     // 48
#define BHC_  (BH_*C_)    // 624
#define M_    (B_*N_*C_)  // 10192 = 637*16
#define MP_   10240       // M padded to multiple of 64 (zero-filled tail rows)
#define QKVN_ (5*D_)      // 3840
#define NPAD_ 224         // 196 padded up to multiple of 32 (zero-filled)
#define SCALE_ 0.125f     // 64^-0.5

// ---------------------------------------------------------------------------
// WMMA helpers. Fragment layout per CDNA5 ISA (16-bit A-matrix 16x32 table):
// lane L holds row (L&15); lo lanes (L<16): elems 0..7 = K kb+0..7,
// elems 8..15 = K kb+16..23; hi lanes: +8 shift. B is loaded identically from
// B^T stored row-major [n][k]. Two 16-byte v8bf loads per fragment per lane;
// all row strides are multiples of 16 elements so loads stay 16-B aligned.
// ---------------------------------------------------------------------------
__device__ __forceinline__ v16bf load_frag_bf16(const bf16_t* __restrict__ base,
                                                int row, int ld, int kbase, int lane) {
    const bf16_t* p = base + (long)row * ld + kbase + ((lane >> 4) << 3);
    union { v16bf v; v8bf h[2]; } u;
    u.h[0] = *(const v8bf*)p;
    u.h[1] = *(const v8bf*)(p + 16);
    return u.v;
}

__device__ __forceinline__ v8f wmma_bf16(v16bf a, v16bf b, v8f c) {
    // D = A*B + C, f32 accumulate
    return __builtin_amdgcn_wmma_f32_16x16x32_bf16(false, a, false, b,
                                                   (short)0, c, false, false);
}

// ---------------------------------------------------------------------------
// K0: fp32 -> bf16 convert / zero-fill
// ---------------------------------------------------------------------------
__global__ void k_cvt(const float* __restrict__ in, bf16_t* __restrict__ out, long n) {
    long i = (long)blockIdx.x * blockDim.x + threadIdx.x;
    long s = (long)gridDim.x * blockDim.x;
    for (; i < n; i += s) out[i] = (bf16_t)in[i];
}

__global__ void k_zero(bf16_t* __restrict__ p, long n) {
    long i = (long)blockIdx.x * blockDim.x + threadIdx.x;
    long s = (long)gridDim.x * blockDim.x;
    for (; i < n; i += s) p[i] = (bf16_t)0.0f;
}

// ---------------------------------------------------------------------------
// K1: QKV GEMM (10240x768 @ 3840x768^T) with fused 5-way scatter + q scaling.
// Each wave computes a 64x64 block (4x4 WMMA tiles, 16 f32 accumulators):
// each K-step loads 8 fragments for 16 WMMAs (0.5 frag/WMMA vs 2.0
// unblocked -> 4x less L2 traffic). 2 waves per 64-thread block, tiled flat.
// N-blocks of 64 never straddle an (s,h) boundary (768%64==0).
// ---------------------------------------------------------------------------
__global__ void __launch_bounds__(64) k_qkv(const bf16_t* __restrict__ Xb,
                                            const bf16_t* __restrict__ Wb,
                                            bf16_t* __restrict__ qc,
                                            bf16_t* __restrict__ kc,
                                            bf16_t* __restrict__ qs,
                                            bf16_t* __restrict__ ks,
                                            bf16_t* __restrict__ vT) {
    const int NB = QKVN_ / 64;  // 60
    int wid = blockIdx.x * 2 + (threadIdx.x >> 5);
    int mb = wid / NB, nb = wid % NB;
    int lane = threadIdx.x & 31;
    int l15 = lane & 15;
    int rbase = mb * 64;
    int ncol0 = nb * 64;
    v8f acc[4][4] = {};
    for (int kb = 0; kb < D_; kb += 32) {
        v16bf a0 = load_frag_bf16(Xb, rbase + 0 * 16 + l15, D_, kb, lane);
        v16bf a1 = load_frag_bf16(Xb, rbase + 1 * 16 + l15, D_, kb, lane);
        v16bf a2 = load_frag_bf16(Xb, rbase + 2 * 16 + l15, D_, kb, lane);
        v16bf a3 = load_frag_bf16(Xb, rbase + 3 * 16 + l15, D_, kb, lane);
        v16bf b0 = load_frag_bf16(Wb, ncol0 + 0 * 16 + l15, D_, kb, lane);
        v16bf b1 = load_frag_bf16(Wb, ncol0 + 1 * 16 + l15, D_, kb, lane);
        v16bf b2 = load_frag_bf16(Wb, ncol0 + 2 * 16 + l15, D_, kb, lane);
        v16bf b3 = load_frag_bf16(Wb, ncol0 + 3 * 16 + l15, D_, kb, lane);
        acc[0][0] = wmma_bf16(a0, b0, acc[0][0]);
        acc[0][1] = wmma_bf16(a0, b1, acc[0][1]);
        acc[0][2] = wmma_bf16(a0, b2, acc[0][2]);
        acc[0][3] = wmma_bf16(a0, b3, acc[0][3]);
        acc[1][0] = wmma_bf16(a1, b0, acc[1][0]);
        acc[1][1] = wmma_bf16(a1, b1, acc[1][1]);
        acc[1][2] = wmma_bf16(a1, b2, acc[1][2]);
        acc[1][3] = wmma_bf16(a1, b3, acc[1][3]);
        acc[2][0] = wmma_bf16(a2, b0, acc[2][0]);
        acc[2][1] = wmma_bf16(a2, b1, acc[2][1]);
        acc[2][2] = wmma_bf16(a2, b2, acc[2][2]);
        acc[2][3] = wmma_bf16(a2, b3, acc[2][3]);
        acc[3][0] = wmma_bf16(a3, b0, acc[3][0]);
        acc[3][1] = wmma_bf16(a3, b1, acc[3][1]);
        acc[3][2] = wmma_bf16(a3, b2, acc[3][2]);
        acc[3][3] = wmma_bf16(a3, b3, acc[3][3]);
    }
    int s = ncol0 / D_;               // which of qc/kc/qs/ks/v (wave-uniform)
    int h = (ncol0 % D_) / HD_;       // head (wave-uniform)
    int hi8 = (lane >> 4) << 3;
    for (int i = 0; i < 4; ++i) {
        for (int j = 0; j < 4; ++j) {
            int t = j * 16 + l15;     // 0..63 within head
            for (int e = 0; e < 8; ++e) {
                int m = rbase + i * 16 + e + hi8;
                if (m >= M_) continue;
                int b0i = m / (N_ * C_), rem = m % (N_ * C_);
                int nn = rem / C_, c = rem % C_;
                float val = acc[i][j][e];
                long bh = (long)b0i * H_ + h;
                if (s == 0)      qc[(((bh * N_ + nn) * C_ + c) * HD_) + t] = (bf16_t)(val * SCALE_);
                else if (s == 1) kc[(((bh * N_ + nn) * C_ + c) * HD_) + t] = (bf16_t)val;
                else if (s == 2) qs[(((bh * C_ + c) * N_ + nn) * HD_) + t] = (bf16_t)(val * SCALE_);
                else if (s == 3) ks[(((bh * C_ + c) * N_ + nn) * HD_) + t] = (bf16_t)val;
                else             vT[(((bh * C_ + c) * HD_ + t) * NPAD_) + nn] = (bf16_t)val;
            }
        }
    }
}

// ---------------------------------------------------------------------------
// K2: spatial attention. One wave per (b,h,c, 16-row tile of queries):
// WMMA 196x196 scores (K=64, Q fragments hoisted out of the column loop),
// row softmax in LDS, write bf16 rows zero-padded to 224 so downstream S@V
// needs no K masking.
// ---------------------------------------------------------------------------
__global__ void __launch_bounds__(32) k_spatial(const bf16_t* __restrict__ qs,
                                                const bf16_t* __restrict__ ks,
                                                bf16_t* __restrict__ attn_s) {
    __shared__ float sc[16 * NPAD_];
    int bhc = blockIdx.x / 13;
    int mt  = blockIdx.x % 13;
    int lane = threadIdx.x;
    const bf16_t* Q  = qs + (long)bhc * N_ * HD_;
    const bf16_t* Kp = ks + (long)bhc * N_ * HD_;
    int mrow = mt * 16 + (lane & 15); if (mrow > N_ - 1) mrow = N_ - 1;
    int hi8 = (lane >> 4) << 3;
    v16bf aq0 = load_frag_bf16(Q, mrow, HD_, 0,  lane);
    v16bf aq1 = load_frag_bf16(Q, mrow, HD_, 32, lane);
    for (int ntile = 0; ntile < 13; ++ntile) {
        int ncol = ntile * 16 + (lane & 15);
        int nrow = (ncol > N_ - 1) ? (N_ - 1) : ncol;
        v16bf b0 = load_frag_bf16(Kp, nrow, HD_, 0,  lane);
        v16bf b1 = load_frag_bf16(Kp, nrow, HD_, 32, lane);
        v8f acc = {};
        acc = wmma_bf16(aq0, b0, acc);
        acc = wmma_bf16(aq1, b1, acc);
        for (int e = 0; e < 8; ++e)
            sc[(e + hi8) * NPAD_ + ncol] = (ncol < N_) ? acc[e] : 0.0f;
    }
    for (int i = lane; i < 16 * 16; i += 32)               // zero cols 208..223
        sc[(i >> 4) * NPAD_ + 208 + (i & 15)] = 0.0f;
    __syncthreads();
    for (int rr = 0; rr < 16; ++rr) {
        int x = mt * 16 + rr;
        if (x >= N_) break;
        float mx = -3.0e38f;
        for (int c2 = lane; c2 < N_; c2 += 32) mx = fmaxf(mx, sc[rr * NPAD_ + c2]);
        for (int o = 16; o; o >>= 1) mx = fmaxf(mx, __shfl_xor(mx, o, 32));
        float sum = 0.0f;
        for (int c2 = lane; c2 < N_; c2 += 32) {
            float ev = __expf(sc[rr * NPAD_ + c2] - mx);
            sc[rr * NPAD_ + c2] = ev;
            sum += ev;
        }
        for (int o = 16; o; o >>= 1) sum += __shfl_xor(sum, o, 32);
        float inv = 1.0f / sum;
        for (int c2 = lane; c2 < NPAD_; c2 += 32) {
            float v = (c2 < N_) ? sc[rr * NPAD_ + c2] * inv : 0.0f;
            attn_s[((long)bhc * N_ + x) * NPAD_ + c2] = (bf16_t)v;
        }
    }
}

// ---------------------------------------------------------------------------
// K3: channel attention (13x13 softmax per (b,h,n)) — small, VALU + LDS.
// qc was pre-scaled by SCALE in K1.
// ---------------------------------------------------------------------------
__global__ void __launch_bounds__(192) k_channel(const bf16_t* __restrict__ qc,
                                                 const bf16_t* __restrict__ kc,
                                                 float* __restrict__ attn_c) {
    __shared__ float q[C_ * HD_], k[C_ * HD_], sc[C_ * C_];
    long base = (long)blockIdx.x * (C_ * HD_);
    int t = threadIdx.x;
    for (int i = t; i < C_ * HD_; i += blockDim.x) {
        q[i] = (float)qc[base + i];
        k[i] = (float)kc[base + i];
    }
    __syncthreads();
    if (t < C_ * C_) {
        int qi = t / C_, ki = t % C_;
        float s = 0.0f;
        for (int d = 0; d < HD_; ++d) s += q[qi * HD_ + d] * k[ki * HD_ + d];
        sc[t] = s;
    }
    __syncthreads();
    if (t < C_) {
        float mx = -3.0e38f;
        for (int ki = 0; ki < C_; ++ki) mx = fmaxf(mx, sc[t * C_ + ki]);
        float ev[C_], sum = 0.0f;
        for (int ki = 0; ki < C_; ++ki) { ev[ki] = __expf(sc[t * C_ + ki] - mx); sum += ev[ki]; }
        float inv = 1.0f / sum;
        long ob = (long)blockIdx.x * (C_ * C_) + (long)t * C_;
        for (int ki = 0; ki < C_; ++ki) attn_c[ob + ki] = ev[ki] * inv;
    }
}

// ---------------------------------------------------------------------------
// K4: fused  out[x,y,:] = sum_c attn_c[x,y,c] * (S_y @ V_c)[x,:]
// One wave per (b,h,y, 16-row x-tile). All 7 K-chunk A-fragments of S_y are
// hoisted into registers (56 VGPRs) and reused across the 13 c-GEMMs.
// ---------------------------------------------------------------------------
__global__ void __launch_bounds__(32) k_svcombine(const bf16_t* __restrict__ attn_s,
                                                  const bf16_t* __restrict__ vT,
                                                  const float* __restrict__ attn_c,
                                                  bf16_t* __restrict__ outpre) {
    int bhy = blockIdx.x / 13;
    int xt  = blockIdx.x % 13;
    int bh = bhy / C_, y = bhy % C_;
    int lane = threadIdx.x;
    const bf16_t* S = attn_s + ((long)bh * C_ + y) * N_ * NPAD_;
    int mrow = xt * 16 + (lane & 15);
    int mclamp = (mrow > N_ - 1) ? (N_ - 1) : mrow;
    int hi8 = (lane >> 4) << 3;
    v16bf af[NPAD_ / 32];
    for (int kk = 0; kk < NPAD_ / 32; ++kk)
        af[kk] = load_frag_bf16(S, mclamp, NPAD_, kk * 32, lane);
    v8f r0 = {}, r1 = {}, r2 = {}, r3 = {};
    for (int c = 0; c < C_; ++c) {
        const bf16_t* Vt = vT + ((long)bh * C_ + c) * HD_ * NPAD_;
        v8f p0 = {}, p1 = {}, p2 = {}, p3 = {};
        for (int kk = 0; kk < NPAD_ / 32; ++kk) {
            int kb = kk * 32;
            v16bf b0 = load_frag_bf16(Vt, 0 * 16 + (lane & 15), NPAD_, kb, lane);
            v16bf b1 = load_frag_bf16(Vt, 1 * 16 + (lane & 15), NPAD_, kb, lane);
            v16bf b2 = load_frag_bf16(Vt, 2 * 16 + (lane & 15), NPAD_, kb, lane);
            v16bf b3 = load_frag_bf16(Vt, 3 * 16 + (lane & 15), NPAD_, kb, lane);
            p0 = wmma_bf16(af[kk], b0, p0);
            p1 = wmma_bf16(af[kk], b1, p1);
            p2 = wmma_bf16(af[kk], b2, p2);
            p3 = wmma_bf16(af[kk], b3, p3);
        }
        float wr[8];
        for (int e = 0; e < 8; ++e) {
            int x = xt * 16 + e + hi8;
            wr[e] = (x < N_) ? attn_c[((long)bh * N_ + x) * (C_ * C_) + y * C_ + c] : 0.0f;
        }
        for (int e = 0; e < 8; ++e) {
            r0[e] += wr[e] * p0[e];
            r1[e] += wr[e] * p1[e];
            r2[e] += wr[e] * p2[e];
            r3[e] += wr[e] * p3[e];
        }
    }
    int b0_ = bh / H_, h = bh % H_;
    int dcol = lane & 15;
    for (int e = 0; e < 8; ++e) {
        int x = xt * 16 + e + hi8;
        if (x < N_) {
            long ob = (((long)b0_ * N_ + x) * C_ + y) * D_ + h * HD_;
            outpre[ob + 0 * 16 + dcol] = (bf16_t)r0[e];
            outpre[ob + 1 * 16 + dcol] = (bf16_t)r1[e];
            outpre[ob + 2 * 16 + dcol] = (bf16_t)r2[e];
            outpre[ob + 3 * 16 + dcol] = (bf16_t)r3[e];
        }
    }
}

// ---------------------------------------------------------------------------
// K5: projection GEMM (10240x768 @ 768x768^T) + bias, fp32 out.
// Same 64x64-per-wave register blocking as K1.
// ---------------------------------------------------------------------------
__global__ void __launch_bounds__(64) k_proj(const bf16_t* __restrict__ Yb,
                                             const bf16_t* __restrict__ Wb,
                                             const float* __restrict__ bias,
                                             float* __restrict__ out) {
    const int NB = D_ / 64;  // 12
    int wid = blockIdx.x * 2 + (threadIdx.x >> 5);
    int mb = wid / NB, nb = wid % NB;
    int lane = threadIdx.x & 31;
    int l15 = lane & 15;
    int rbase = mb * 64;
    int ncol0 = nb * 64;
    v8f acc[4][4] = {};
    for (int kb = 0; kb < D_; kb += 32) {
        v16bf a0 = load_frag_bf16(Yb, rbase + 0 * 16 + l15, D_, kb, lane);
        v16bf a1 = load_frag_bf16(Yb, rbase + 1 * 16 + l15, D_, kb, lane);
        v16bf a2 = load_frag_bf16(Yb, rbase + 2 * 16 + l15, D_, kb, lane);
        v16bf a3 = load_frag_bf16(Yb, rbase + 3 * 16 + l15, D_, kb, lane);
        v16bf b0 = load_frag_bf16(Wb, ncol0 + 0 * 16 + l15, D_, kb, lane);
        v16bf b1 = load_frag_bf16(Wb, ncol0 + 1 * 16 + l15, D_, kb, lane);
        v16bf b2 = load_frag_bf16(Wb, ncol0 + 2 * 16 + l15, D_, kb, lane);
        v16bf b3 = load_frag_bf16(Wb, ncol0 + 3 * 16 + l15, D_, kb, lane);
        acc[0][0] = wmma_bf16(a0, b0, acc[0][0]);
        acc[0][1] = wmma_bf16(a0, b1, acc[0][1]);
        acc[0][2] = wmma_bf16(a0, b2, acc[0][2]);
        acc[0][3] = wmma_bf16(a0, b3, acc[0][3]);
        acc[1][0] = wmma_bf16(a1, b0, acc[1][0]);
        acc[1][1] = wmma_bf16(a1, b1, acc[1][1]);
        acc[1][2] = wmma_bf16(a1, b2, acc[1][2]);
        acc[1][3] = wmma_bf16(a1, b3, acc[1][3]);
        acc[2][0] = wmma_bf16(a2, b0, acc[2][0]);
        acc[2][1] = wmma_bf16(a2, b1, acc[2][1]);
        acc[2][2] = wmma_bf16(a2, b2, acc[2][2]);
        acc[2][3] = wmma_bf16(a2, b3, acc[2][3]);
        acc[3][0] = wmma_bf16(a3, b0, acc[3][0]);
        acc[3][1] = wmma_bf16(a3, b1, acc[3][1]);
        acc[3][2] = wmma_bf16(a3, b2, acc[3][2]);
        acc[3][3] = wmma_bf16(a3, b3, acc[3][3]);
    }
    int hi8 = (lane >> 4) << 3;
    for (int i = 0; i < 4; ++i) {
        for (int j = 0; j < 4; ++j) {
            int ncol = ncol0 + j * 16 + l15;
            float bn = bias[ncol];
            for (int e = 0; e < 8; ++e) {
                int m = rbase + i * 16 + e + hi8;
                if (m < M_) out[(long)m * D_ + ncol] = acc[i][j][e] + bn;
            }
        }
    }
}

// ---------------------------------------------------------------------------
extern "C" void kernel_launch(void* const* d_in, const int* in_sizes, int n_in,
                              void* d_out, int out_size, void* d_ws, size_t ws_size,
                              hipStream_t stream) {
    const float* x     = (const float*)d_in[0];
    const float* Wqkv  = (const float*)d_in[1];
    const float* Wproj = (const float*)d_in[2];
    const float* bproj = (const float*)d_in[3];
    float* out = (float*)d_out;

    char* ws = (char*)d_ws;
    size_t off = 0;
    auto take = [&](size_t bytes) -> char* {
        char* p = ws + off;
        off += (bytes + 255) & ~(size_t)255;
        return p;
    };
    bf16_t* Xb     = (bf16_t*)take((size_t)MP_ * D_ * 2);
    bf16_t* Wqb    = (bf16_t*)take((size_t)QKVN_ * D_ * 2);
    bf16_t* Wpb    = (bf16_t*)take((size_t)D_ * D_ * 2);
    bf16_t* qc     = (bf16_t*)take((size_t)BH_ * N_ * C_ * HD_ * 2);
    bf16_t* kc     = (bf16_t*)take((size_t)BH_ * N_ * C_ * HD_ * 2);
    bf16_t* qsb    = (bf16_t*)take((size_t)BHC_ * N_ * HD_ * 2);
    bf16_t* ksb    = (bf16_t*)take((size_t)BHC_ * N_ * HD_ * 2);
    bf16_t* vT     = (bf16_t*)take((size_t)BHC_ * HD_ * NPAD_ * 2);
    bf16_t* attn_s = (bf16_t*)take((size_t)BHC_ * N_ * NPAD_ * 2);
    float*  attn_c = (float*)take((size_t)BH_ * N_ * C_ * C_ * 4);
    bf16_t* outpre = (bf16_t*)take((size_t)MP_ * D_ * 2);

    // converts + zero pads (pad rows of Xb/outpre, padded cols of vT)
    k_cvt<<<2048, 256, 0, stream>>>(x, Xb, (long)M_ * D_);
    k_zero<<<64, 256, 0, stream>>>(Xb + (size_t)M_ * D_, (long)(MP_ - M_) * D_);
    k_cvt<<<2048, 256, 0, stream>>>(Wqkv, Wqb, (long)QKVN_ * D_);
    k_cvt<<<1024, 256, 0, stream>>>(Wproj, Wpb, (long)D_ * D_);
    k_zero<<<2048, 256, 0, stream>>>(vT, (long)BHC_ * HD_ * NPAD_);
    k_zero<<<64, 256, 0, stream>>>(outpre + (size_t)M_ * D_, (long)(MP_ - M_) * D_);

    k_qkv<<<(MP_ / 64) * (QKVN_ / 64) / 2, 64, 0, stream>>>(Xb, Wqb, qc, kc, qsb, ksb, vT);
    k_spatial<<<BHC_ * 13, 32, 0, stream>>>(qsb, ksb, attn_s);
    k_channel<<<BH_ * N_, 192, 0, stream>>>(qc, kc, attn_c);
    k_svcombine<<<BH_ * C_ * 13, 32, 0, stream>>>(attn_s, vT, attn_c, outpre);
    k_proj<<<(MP_ / 64) * (D_ / 64) / 2, 64, 0, stream>>>(outpre, Wpb, bproj, out);
}
